// DeepONetsWithFNO_39487929319466
// MI455X (gfx1250) — matrix operand order repaired and checked by reading, not compiled, not run on hardware
//
#include <hip/hip_runtime.h>
#include <math.h>

#define NEG_SLOPE 0.1f
#define PI_F 3.14159265358979323846f
#define APITCH 264   // 256 + 8 halfs of padding per activation row

typedef __bf16 v16bf __attribute__((ext_vector_type(16)));
typedef __bf16 v8bf  __attribute__((ext_vector_type(8)));
typedef float  v8f   __attribute__((ext_vector_type(8)));
typedef unsigned int tdm_u32x4 __attribute__((ext_vector_type(4)));
typedef int          tdm_i32x8 __attribute__((ext_vector_type(8)));
typedef int          tdm_i32x4 __attribute__((ext_vector_type(4)));

__device__ __forceinline__ float leaky(float x) { return x > 0.0f ? x : NEG_SLOPE * x; }

// ---------------------------------------------------------------------------
// Branch stage 1: X = DFT_1024(parameter)   (naive DFT w/ LDS twiddle table)
// grid 64, block 256
// ---------------------------------------------------------------------------
__global__ __launch_bounds__(256) void dft1024_kernel(const float* __restrict__ x,
                                                      float* __restrict__ Xre,
                                                      float* __restrict__ Xim) {
  __shared__ float tb[1024];
  __shared__ float xs[1024];
  const int b = blockIdx.x;
  const int t = threadIdx.x;
  for (int i = t; i < 1024; i += 256) {
    tb[i] = cosf(2.0f * PI_F * (float)i / 1024.0f);
    xs[i] = x[b * 1024 + i];
  }
  __syncthreads();
  for (int mi = 0; mi < 4; ++mi) {
    const int m = t + mi * 256;
    float re = 0.0f, im = 0.0f;
    int idx = 0;                          // idx = j*m accumulated
    for (int j = 0; j < 1024; ++j) {
      const float c = tb[idx & 1023];               // cos(2*pi*jm/1024)
      const float s = tb[(idx + 768) & 1023];       // sin(2*pi*jm/1024)
      const float xv = xs[j];
      re = __builtin_fmaf(xv, c, re);               // e^{-i theta}
      im = __builtin_fmaf(xv, -s, im);
      idx += m;
    }
    Xre[b * 1024 + m] = re;
    Xim[b * 1024 + m] = im;
  }
}

// ---------------------------------------------------------------------------
// Branch stage 2: Y = X @ W_branch   (complex x real)
// grid 64, block 256 (one output column per thread)
// ---------------------------------------------------------------------------
__global__ __launch_bounds__(256) void spectral_mix_kernel(const float* __restrict__ Xre,
                                                           const float* __restrict__ Xim,
                                                           const float* __restrict__ Wb,
                                                           float* __restrict__ Yre,
                                                           float* __restrict__ Yim) {
  const int b = blockIdx.x;
  const int n = threadIdx.x;
  float re = 0.0f, im = 0.0f;
  for (int m = 0; m < 1024; ++m) {
    const float w = Wb[m * 256 + n];
    re = __builtin_fmaf(Xre[b * 1024 + m], w, re);
    im = __builtin_fmaf(Xim[b * 1024 + m], w, im);
  }
  Yre[b * 256 + n] = re;
  Yim[b * 256 + n] = im;
}

// ---------------------------------------------------------------------------
// Branch stage 3: branch = Re(IDFT_256(Y)); fold Wo:  v[b,k] = branch[b,k]*Wo[k]
// grid 64, block 256
// ---------------------------------------------------------------------------
__global__ __launch_bounds__(256) void ifft_fold_kernel(const float* __restrict__ Yre,
                                                        const float* __restrict__ Yim,
                                                        const float* __restrict__ Wo,
                                                        float* __restrict__ vbr) {
  __shared__ float tb[256];
  __shared__ float yr[256];
  __shared__ float yi[256];
  const int b = blockIdx.x;
  const int k = threadIdx.x;
  tb[k] = cosf(2.0f * PI_F * (float)k / 256.0f);
  yr[k] = Yre[b * 256 + k];
  yi[k] = Yim[b * 256 + k];
  __syncthreads();
  float acc = 0.0f;
  int idx = 0;                            // idx = n*k accumulated
  for (int n = 0; n < 256; ++n) {
    const float c = tb[idx & 255];                  // cos(2*pi*nk/256)
    const float s = tb[(idx + 192) & 255];          // sin(2*pi*nk/256)
    acc += yr[n] * c - yi[n] * s;                   // Re{Y * e^{+i theta}}
    idx += k;
  }
  vbr[b * 256 + k] = acc * (1.0f / 256.0f) * Wo[k];
}

// ---------------------------------------------------------------------------
// Weight prep: W2/W3 (f32 row-major [K][N]) -> bf16 fragment-swizzled layout.
// For block (kt 0..7, ct 0..15): lane l holds B[kt*32 + (l>=16?16:0) + i][ct*16 + l%16]
// stored at gW[layer*65536 + ((kt*16+ct)*32 + l)*16 + i]  (contiguous per wave).
// grid 512, block 256  (131072 bf16 elements total)
// ---------------------------------------------------------------------------
__global__ __launch_bounds__(256) void swizzle_weights_kernel(const float* __restrict__ W2,
                                                              const float* __restrict__ W3,
                                                              __bf16* __restrict__ gW) {
  const int e = blockIdx.x * 256 + threadIdx.x;     // [0, 131072)
  const float* W = (e >> 16) ? W3 : W2;
  const int r   = e & 65535;
  const int i   = r & 15;
  const int l   = (r >> 4) & 31;
  const int blk = r >> 9;                            // 0..127
  const int ct  = blk & 15;
  const int kt  = blk >> 4;                          // 0..7
  const int n   = ct * 16 + (l & 15);
  const int k   = kt * 32 + ((l >> 4) << 4) + i;
  gW[e] = (__bf16)W[k * 256 + n];
}

// ---------------------------------------------------------------------------
// Fused trunk: per block = 64 rows of flattened [B*P].
//   layer1 (K=4) in VALU -> bf16 LDS acts
//   layer2/layer3: bf16 WMMA 16x16x32; weights staged into LDS in 64KB K-halves
//   via the Tensor Data Mover (TENSOR_LOAD_TO_LDS, TENSORcnt-tracked)
//   wave tile = 32 rows x 64 cols (2 row-tiles x 4 col-tiles): every B fragment
//   feeds two WMMAs, halving B-side LDS bandwidth
//   final: leaky, multiply by v[b,:], reduce over H, + bo
// grid 4096, block 256 (8 wave32)
// ---------------------------------------------------------------------------
__global__ __launch_bounds__(256) void trunk_fused_kernel(
    const float* __restrict__ coords, const float* __restrict__ W1,
    const float* __restrict__ b1, const float* __restrict__ b2,
    const float* __restrict__ b3, const __bf16* __restrict__ gW,
    const float* __restrict__ vbr, const float* __restrict__ bo,
    float* __restrict__ out) {
  __shared__ __attribute__((aligned(32))) __bf16 sW[32768];        // 64 KB weight stage
  __shared__ __attribute__((aligned(32))) __bf16 sA[64 * APITCH];  // 33 KB acts ping
  __shared__ __attribute__((aligned(32))) __bf16 sB[64 * APITCH];  // 33 KB acts pong
  __shared__ float rowsum[64];

  const int tid = threadIdx.x;
  const long rowBase = (long)blockIdx.x * 64;
  const int bidx = (int)(rowBase >> 12);             // batch index (P = 4096)

  if (tid < 64) rowsum[tid] = bo[0];

  // ---- layer 1: [64,4] @ [4,256] + b1, leaky -> bf16 in sA (4 threads/row) ----
  {
    const int r  = tid >> 2;
    const int cq = (tid & 3) * 64;
    const float4 cc = ((const float4*)coords)[rowBase + r];
    #pragma unroll 8
    for (int c = cq; c < cq + 64; ++c) {
      float a = b1[c];
      a = __builtin_fmaf(cc.x, W1[c],        a);
      a = __builtin_fmaf(cc.y, W1[256 + c],  a);
      a = __builtin_fmaf(cc.z, W1[512 + c],  a);
      a = __builtin_fmaf(cc.w, W1[768 + c],  a);
      sA[r * APITCH + c] = (__bf16)leaky(a);
    }
    // warm GL2 with this layer-2/3 weight working set while layer 1 runs
    __builtin_prefetch(gW + tid * 256, 0, 1);           // first 128 KB (W2)
    __builtin_prefetch(gW + 65536 + tid * 256, 0, 1);   // second 128 KB (W3)
  }
  __syncthreads();

  const int lane = tid & 31;
  const int wave = tid >> 5;
  const int rh   = wave >> 2;   // row group 0..1 (32 rows each)
  const int cq   = wave & 3;    // column quarter (64 cols)
  const int lm   = lane & 15;
  const int lh   = lane >> 4;

  const unsigned ldsW = (unsigned)(unsigned long long)(void*)sW;  // LDS byte offset of sW

  const __bf16* actIn = sA;
  __bf16* actOut = sB;

  v8f acc[2][4];
  for (int layer = 0; layer < 2; ++layer) {
    #pragma unroll
    for (int rt2 = 0; rt2 < 2; ++rt2)
      #pragma unroll
      for (int ct = 0; ct < 4; ++ct)
        #pragma unroll
        for (int q = 0; q < 8; ++q) acc[rt2][ct][q] = 0.0f;

    const __bf16* gWl = gW + layer * 65536;
    for (int kh = 0; kh < 2; ++kh) {
      // ---- TDM: DMA one 64 KB K-half of swizzled weights into sW ----------
      if (wave == 0) {
        const unsigned long long ga =
            (unsigned long long)(const void*)(gWl + kh * 32768);
        tdm_u32x4 g0;
        g0[0] = 1u;                                   // count=1, user descriptor
        g0[1] = ldsW;                                 // lds_addr (bytes)
        g0[2] = (unsigned)ga;                         // global_addr[31:0]
        g0[3] = (unsigned)((ga >> 32) & 0x01FFFFFFu)  // global_addr[56:32]
                | 0x80000000u;                        // type=2 ("image")
        tdm_i32x8 g1;
        g1[0] = 0x00030000;   // workgroup_mask=0, data_size=3 (8B elements)
        g1[1] = 0x20000000;   // tensor_dim0 = 8192 (x8B = 64KB)
        g1[2] = 0x00010000;   // tensor_dim1 = 1
        g1[3] = 0x20000000;   // tile_dim0 = 8192
        g1[4] = 0x00000001;   // tile_dim1 = 1, tile_dim2 = 0
        g1[5] = 0x00002000;   // tensor_dim0_stride = 8192
        g1[6] = 0x20000000;   // tensor_dim1_stride = 8192 (low 16 in [31:16])
        g1[7] = 0x00000000;
        tdm_i32x4 gz4;
        gz4[0] = 0; gz4[1] = 0; gz4[2] = 0; gz4[3] = 0;  // <=2D: groups 2/3 unused
        tdm_i32x8 gz8;
        gz8[0] = 0; gz8[1] = 0; gz8[2] = 0; gz8[3] = 0;
        gz8[4] = 0; gz8[5] = 0; gz8[6] = 0; gz8[7] = 0;
        __builtin_amdgcn_tensor_load_to_lds(g0, g1, gz4, gz4, gz8, 0);
        __builtin_amdgcn_s_wait_tensorcnt(0);
      }
      __syncthreads();

      // ---- 4 K-steps: per step load 2 A frags + 4 B frags, run 8 WMMAs ----
      #pragma unroll
      for (int ktl = 0; ktl < 4; ++ktl) {
        const int kt = kh * 4 + ktl;
        v16bf afr2[2];
        #pragma unroll
        for (int rt2 = 0; rt2 < 2; ++rt2) {
          const __bf16* ap = actIn + (rh * 32 + rt2 * 16 + lm) * APITCH
                           + kt * 32 + lh * 16;
          const v8bf alo = *(const v8bf*)ap;
          const v8bf ahi = *(const v8bf*)(ap + 8);
          afr2[rt2] = __builtin_shufflevector(alo, ahi,
              0,1,2,3,4,5,6,7,8,9,10,11,12,13,14,15);
        }
        const __bf16* bbase = sW + (ktl * 16 + cq * 4) * 512 + lane * 16;
        v16bf bf[4];
        #pragma unroll
        for (int ct = 0; ct < 4; ++ct)
          bf[ct] = *(const v16bf*)(bbase + ct * 512);
        #pragma unroll
        for (int ct = 0; ct < 4; ++ct) {
          acc[0][ct] = __builtin_amdgcn_wmma_f32_16x16x32_bf16(
              false, afr2[0], false, bf[ct], (short)0, acc[0][ct], false, false);
          acc[1][ct] = __builtin_amdgcn_wmma_f32_16x16x32_bf16(
              false, afr2[1], false, bf[ct], (short)0, acc[1][ct], false, false);
        }
      }
      // scheduling hint: keep the next K-step's 12 DS reads in flight ahead of
      // each 8-WMMA group (one-stage software pipeline). Best-effort.
      __builtin_amdgcn_sched_group_barrier(0x100, 12, 0);  // DS reads, step 0
      __builtin_amdgcn_sched_group_barrier(0x100, 12, 0);  // DS reads, step 1
      __builtin_amdgcn_sched_group_barrier(0x008, 8, 0);   // WMMAs,   step 0
      __builtin_amdgcn_sched_group_barrier(0x100, 12, 0);  // DS reads, step 2
      __builtin_amdgcn_sched_group_barrier(0x008, 8, 0);   // WMMAs,   step 1
      __builtin_amdgcn_sched_group_barrier(0x100, 12, 0);  // DS reads, step 3
      __builtin_amdgcn_sched_group_barrier(0x008, 8, 0);   // WMMAs,   step 2
      __builtin_amdgcn_sched_group_barrier(0x008, 8, 0);   // WMMAs,   step 3
      __syncthreads();
    }

    if (layer == 0) {
      // bias + leaky -> bf16 acts for next layer
      #pragma unroll
      for (int rt2 = 0; rt2 < 2; ++rt2) {
        #pragma unroll
        for (int ct = 0; ct < 4; ++ct) {
          const int col = cq * 64 + ct * 16 + lm;
          const float bb = b2[col];
          const int rb = rh * 32 + rt2 * 16 + lh * 8;
          #pragma unroll
          for (int r = 0; r < 8; ++r)
            actOut[(rb + r) * APITCH + col] = (__bf16)leaky(acc[rt2][ct][r] + bb);
        }
      }
      __syncthreads();
      actIn = sB;
      actOut = sA;
    } else {
      // final: t3 = leaky(acc + b3); partial = t3 * v[b,col]; reduce over col
      float rpart[2][8];
      #pragma unroll
      for (int rt2 = 0; rt2 < 2; ++rt2)
        #pragma unroll
        for (int r = 0; r < 8; ++r) rpart[rt2][r] = 0.0f;
      #pragma unroll
      for (int ct = 0; ct < 4; ++ct) {
        const int col = cq * 64 + ct * 16 + lm;
        const float bb = b3[col];
        const float vv = vbr[bidx * 256 + col];
        #pragma unroll
        for (int rt2 = 0; rt2 < 2; ++rt2)
          #pragma unroll
          for (int r = 0; r < 8; ++r)
            rpart[rt2][r] = __builtin_fmaf(leaky(acc[rt2][ct][r] + bb), vv,
                                           rpart[rt2][r]);
      }
      // reduce across the 16 lanes sharing each row group
      #pragma unroll
      for (int m = 8; m >= 1; m >>= 1) {
        #pragma unroll
        for (int rt2 = 0; rt2 < 2; ++rt2)
          #pragma unroll
          for (int r = 0; r < 8; ++r)
            rpart[rt2][r] += __shfl_xor(rpart[rt2][r], m, 32);
      }
      if (lm == 0) {  // lanes 0 and 16 commit their rows (x2 row tiles)
        #pragma unroll
        for (int rt2 = 0; rt2 < 2; ++rt2) {
          const int rb = rh * 32 + rt2 * 16 + lh * 8;
          #pragma unroll
          for (int r = 0; r < 8; ++r) atomicAdd(&rowsum[rb + r], rpart[rt2][r]);
        }
      }
      __syncthreads();
      if (tid < 64) out[rowBase + tid] = rowsum[tid];
    }
  }
}

// ---------------------------------------------------------------------------
extern "C" void kernel_launch(void* const* d_in, const int* in_sizes, int n_in,
                              void* d_out, int out_size, void* d_ws, size_t ws_size,
                              hipStream_t stream) {
  const float* parameter = (const float*)d_in[0];   // [64,1024]
  const float* coords    = (const float*)d_in[1];   // [64,4096,4]
  const float* W_branch  = (const float*)d_in[2];   // [1024,256]
  const float* W1        = (const float*)d_in[3];   // [4,256]
  const float* b1        = (const float*)d_in[4];   // [256]
  const float* W2        = (const float*)d_in[5];   // [256,256]
  const float* b2        = (const float*)d_in[6];   // [256]
  const float* W3        = (const float*)d_in[7];   // [256,256]
  const float* b3        = (const float*)d_in[8];   // [256]
  const float* Wo        = (const float*)d_in[9];   // [256,1]
  const float* bo        = (const float*)d_in[10];  // [1]
  (void)in_sizes; (void)n_in; (void)out_size; (void)ws_size;

  char* ws = (char*)d_ws;
  float*  Xre = (float*)(ws + 0);         // 256 KB
  float*  Xim = (float*)(ws + 262144);    // 256 KB
  float*  Yre = (float*)(ws + 524288);    //  64 KB
  float*  Yim = (float*)(ws + 589824);    //  64 KB
  float*  vbr = (float*)(ws + 655360);    //  64 KB  (branch * Wo, per batch)
  __bf16* gW  = (__bf16*)(ws + 720896);   // 256 KB  (swizzled bf16 W2|W3)

  dft1024_kernel<<<64, 256, 0, stream>>>(parameter, Xre, Xim);
  spectral_mix_kernel<<<64, 256, 0, stream>>>(Xre, Xim, W_branch, Yre, Yim);
  ifft_fold_kernel<<<64, 256, 0, stream>>>(Yre, Yim, Wo, vbr);
  swizzle_weights_kernel<<<512, 256, 0, stream>>>(W2, W3, gW);
  trunk_fused_kernel<<<4096, 256, 0, stream>>>(coords, W1, b1, b2, b3, gW, vbr, bo,
                                               (float*)d_out);
}
